// HybridLoss_8469675508203
// MI455X (gfx1250) — compile-verified, matrix operand order checked by reading, not run on hardware
//
#include <hip/hip_runtime.h>
#include <hip/hip_bf16.h>
#include <math.h>

typedef __attribute__((ext_vector_type(2)))  float  v2f;
typedef __attribute__((ext_vector_type(4)))  float  v4f;
typedef __attribute__((ext_vector_type(8)))  float  v8f;
typedef __attribute__((ext_vector_type(8)))  __bf16 v8bf;
typedef __attribute__((ext_vector_type(16))) __bf16 v16bf;

#define B_ROWS 8192
#define D_DIM  512
#define BM 128
#define BN 128
#define BKB 64        // bf16 K-slab per LDS stage (2 WMMA k-steps of 32)
#define LDSB 72       // bf16 row stride: 144 B -> 16B-aligned, conflict-free
#define EPSV 1e-8f

// exp(-d2/8) = exp2(d2 * (-0.125 * log2(e)))
#define EXP_C (-0.18033688011112042f)

// ---------------------------------------------------------------------------
// f32 -> bf16 (round-to-nearest-even), packed; each thread converts 8 floats.
// ---------------------------------------------------------------------------
__device__ __forceinline__ unsigned int pack_bf16(float lo, float hi) {
    unsigned int ul = __float_as_uint(lo);
    ul = (ul + 0x7FFFu + ((ul >> 16) & 1u)) >> 16;
    unsigned int uh = __float_as_uint(hi);
    uh = (uh + 0x7FFFu + ((uh >> 16) & 1u)) >> 16;
    return ul | (uh << 16);
}

__global__ __launch_bounds__(256) void cvt_bf16_kernel(const float* __restrict__ in,
                                                       unsigned int* __restrict__ out) {
    const size_t i = (size_t)blockIdx.x * 256 + threadIdx.x;  // 8 floats each
    const float4* p = (const float4*)in + i * 2;
    const float4 a = p[0];
    const float4 b = p[1];
    uint4 o;
    o.x = pack_bf16(a.x, a.y);
    o.y = pack_bf16(a.z, a.w);
    o.z = pack_bf16(b.x, b.y);
    o.w = pack_bf16(b.z, b.w);
    ((uint4*)out)[i] = o;
}

// ---------------------------------------------------------------------------
// Row squared-norms (exact f32): one wave per row; rows 0..8191 = source,
// 8192..16383 = target.
// ---------------------------------------------------------------------------
__global__ __launch_bounds__(256) void norms_kernel(const float* __restrict__ S,
                                                    const float* __restrict__ T,
                                                    float* __restrict__ x2,
                                                    float* __restrict__ y2) {
    const int wave = threadIdx.x >> 5;
    const int lane = threadIdx.x & 31;
    const int row  = blockIdx.x * 8 + wave;   // 0..16383
    const float* p = (row < B_ROWS) ? &S[(size_t)row * D_DIM]
                                    : &T[(size_t)(row - B_ROWS) * D_DIM];
    float s = 0.0f;
    for (int k = lane; k < D_DIM; k += 32) { float v = p[k]; s += v * v; }
    #pragma unroll
    for (int off = 16; off > 0; off >>= 1) s += __shfl_xor(s, off, 32);
    if (lane == 0) {
        if (row < B_ROWS) x2[row] = s;
        else              y2[row - B_ROWS] = s;
    }
}

// ---------------------------------------------------------------------------
// Target centroid, stage 1: 64 blocks x 128 rows -> per-block column sums
// ---------------------------------------------------------------------------
__global__ __launch_bounds__(512) void centroid_partial_kernel(const float* __restrict__ T,
                                                               float* __restrict__ pc) {
    const int col = threadIdx.x;          // 0..511
    const int r0  = blockIdx.x * 128;
    float s = 0.0f;
    for (int r = 0; r < 128; ++r) s += T[(size_t)(r0 + r) * D_DIM + col];
    pc[blockIdx.x * D_DIM + col] = s;
}

// Stage 2: reduce 64 partials, also compute centroid norm
__global__ __launch_bounds__(512) void centroid_reduce_kernel(const float* __restrict__ pc,
                                                              float* __restrict__ centroid,
                                                              float* __restrict__ cn) {
    __shared__ float red[512];
    const int col = threadIdx.x;
    float s = 0.0f;
    for (int b = 0; b < 64; ++b) s += pc[b * D_DIM + col];
    const float m = s * (1.0f / (float)B_ROWS);
    centroid[col] = m;
    red[col] = m * m;
    __syncthreads();
    #pragma unroll
    for (int off = 256; off > 0; off >>= 1) {
        if (col < off) red[col] += red[col + off];
        __syncthreads();
    }
    if (col == 0) cn[0] = sqrtf(red[0]);
}

// ---------------------------------------------------------------------------
// Fused Gram + RBF-MMD kernel (bf16 WMMA, f32 accumulate).
// grid = (64, 64, 3): z=0 -> XX, z=1 -> YY, z=2 -> XY.
// 256 threads = 8 waves; block tile 128x128; wave tile 32x64 (2x4 WMMA tiles).
// V_WMMA_F32_16X16X32_BF16 over K=512 in LDS slabs of 64 bf16.
// 16-bit A-fragment layout per lane: K = {0..7,16..23} (lanes 0-15) or
// {8..15,24..31} (lanes 16-31) -> two contiguous 16B chunks per k-step.
// Epilogue: d2 = ||x||^2+||y||^2-2*dot (norms in exact f32),
// kernel = (u + u^4 + u^16)/3 with u = exp(-d2/8), diagonal masked for XX/YY.
// ---------------------------------------------------------------------------
__global__ __launch_bounds__(256) void gram_mmd_kernel(const unsigned short* __restrict__ Xbf,
                                                       const unsigned short* __restrict__ Ybf,
                                                       const float* __restrict__ x2,
                                                       const float* __restrict__ y2,
                                                       float* __restrict__ gram_partials) {
    __shared__ unsigned short As[BM * LDSB];
    __shared__ unsigned short Bs[BN * LDSB];
    __shared__ float a2s[BM];
    __shared__ float b2s[BN];
    __shared__ float red[256];

    const int z = blockIdx.z;
    const unsigned short* Amat = (z == 1) ? Ybf : Xbf;
    const unsigned short* Bmat = (z == 0) ? Xbf : Ybf;
    const float* a2 = (z == 1) ? y2 : x2;
    const float* b2 = (z == 0) ? x2 : y2;
    const bool skipDiag = (z != 2);

    const int m0   = blockIdx.x * BM;
    const int n0   = blockIdx.y * BN;
    const int tid  = threadIdx.x;
    const int lane = tid & 31;
    const int wave = tid >> 5;
    const int waveM = wave & 3;    // 4 waves over M: 32-row slices
    const int waveN = wave >> 2;   // 2 waves over N: 64-col slices
    const int halfK = lane >> 4;   // fragment K-half select
    const int r16   = lane & 15;

    if (tid < BM) a2s[tid]      = a2[m0 + tid];
    else          b2s[tid - BM] = b2[n0 + tid - BM];

    v8f acc[2][4];
    const v8f zero8 = {0.f, 0.f, 0.f, 0.f, 0.f, 0.f, 0.f, 0.f};
    #pragma unroll
    for (int i = 0; i < 2; ++i)
        #pragma unroll
        for (int j = 0; j < 4; ++j) acc[i][j] = zero8;

    for (int kk = 0; kk < D_DIM; kk += BKB) {
        // Stage 128x64 bf16 slabs: 1024 x 16B chunks per matrix, 4 per thread.
        #pragma unroll
        for (int it = 0; it < 4; ++it) {
            const int f   = tid + it * 256;   // 0..1023
            const int row = f >> 3;
            const int c   = (f & 7) << 3;     // bf16 offset within row
            const uint4 av = *(const uint4*)&Amat[(size_t)(m0 + row) * D_DIM + kk + c];
            const uint4 bv = *(const uint4*)&Bmat[(size_t)(n0 + row) * D_DIM + kk + c];
            *(uint4*)&As[row * LDSB + c] = av;
            *(uint4*)&Bs[row * LDSB + c] = bv;
        }
        __syncthreads();

        if (kk + BKB < D_DIM) {  // hint next slab into cache (global_prefetch_b8)
            const int row = tid >> 3;
            const int c   = (tid & 7) << 3;
            __builtin_prefetch(&Amat[(size_t)(m0 + row) * D_DIM + kk + BKB + c], 0, 0);
            __builtin_prefetch(&Bmat[(size_t)(n0 + row) * D_DIM + kk + BKB + c], 0, 0);
        }

        #pragma unroll
        for (int ks = 0; ks < 2; ++ks) {      // two K=32 steps per slab
            v16bf af[2], bfr[4];
            #pragma unroll
            for (int i = 0; i < 2; ++i) {
                const int r   = waveM * 32 + i * 16 + r16;
                const int idx = r * LDSB + ks * 32 + halfK * 8;
                const v8bf lo = *(const v8bf*)&As[idx];
                const v8bf hi = *(const v8bf*)&As[idx + 16];
                af[i] = __builtin_shufflevector(lo, hi, 0, 1, 2, 3, 4, 5, 6, 7,
                                                8, 9, 10, 11, 12, 13, 14, 15);
            }
            #pragma unroll
            for (int j = 0; j < 4; ++j) {
                const int r   = waveN * 64 + j * 16 + r16;
                const int idx = r * LDSB + ks * 32 + halfK * 8;
                const v8bf lo = *(const v8bf*)&Bs[idx];
                const v8bf hi = *(const v8bf*)&Bs[idx + 16];
                bfr[j] = __builtin_shufflevector(lo, hi, 0, 1, 2, 3, 4, 5, 6, 7,
                                                 8, 9, 10, 11, 12, 13, 14, 15);
            }
            #pragma unroll
            for (int i = 0; i < 2; ++i)
                #pragma unroll
                for (int j = 0; j < 4; ++j)
                    acc[i][j] = __builtin_amdgcn_wmma_f32_16x16x32_bf16(
                        false, af[i], false, bfr[j], (short)0, acc[i][j],
                        false, false);
        }
        __syncthreads();
    }

    // Epilogue: C/D layout -> lane l, vgpr v holds (M = v + 8*(l/16), N = l%16)
    float s = 0.0f;
    #pragma unroll
    for (int i = 0; i < 2; ++i) {
        const int mloc0 = waveM * 32 + i * 16 + 8 * halfK;
        #pragma unroll
        for (int j = 0; j < 4; ++j) {
            const int nloc = waveN * 64 + j * 16 + r16;
            const float bnorm = b2s[nloc];
            const int n = n0 + nloc;
            #pragma unroll
            for (int v = 0; v < 8; ++v) {
                const int ml = mloc0 + v;
                const int m  = m0 + ml;
                float d2 = a2s[ml] + bnorm - 2.0f * acc[i][j][v];
                d2 = fmaxf(d2, 0.0f);
                const float u   = exp2f(d2 * EXP_C);  // exp(-d2/8)
                const float u2  = u * u;
                const float u4  = u2 * u2;
                const float u8  = u4 * u4;
                const float u16 = u8 * u8;
                float kv = u + u4 + u16;              // sum over 3 bandwidths
                if (skipDiag && m == n) kv = 0.0f;
                s += kv;
            }
        }
    }

    red[tid] = s;
    __syncthreads();
    #pragma unroll
    for (int off = 128; off > 0; off >>= 1) {
        if (tid < off) red[tid] += red[tid + off];
        __syncthreads();
    }
    if (tid == 0)
        gram_partials[z * 4096 + blockIdx.y * 64 + blockIdx.x] = red[0];
}

// ---------------------------------------------------------------------------
// Cosine term (exact f32): one wave per source row; per-block partial sums.
// ---------------------------------------------------------------------------
__global__ __launch_bounds__(256) void cosine_kernel(const float* __restrict__ S,
                                                     const float* __restrict__ centroid,
                                                     const float* __restrict__ x2,
                                                     const float* __restrict__ cnp,
                                                     float* __restrict__ cosp) {
    __shared__ float wsum[8];
    const int wave = threadIdx.x >> 5;
    const int lane = threadIdx.x & 31;
    const int row  = blockIdx.x * 8 + wave;
    const float* p = &S[(size_t)row * D_DIM];
    float s = 0.0f;
    for (int k = lane; k < D_DIM; k += 32) s += p[k] * centroid[k];
    #pragma unroll
    for (int off = 16; off > 0; off >>= 1) s += __shfl_xor(s, off, 32);
    if (lane == 0) {
        const float sn = fmaxf(sqrtf(x2[row]), EPSV);
        const float cn = fmaxf(cnp[0], EPSV);
        wsum[wave] = s / (sn * cn);
    }
    __syncthreads();
    if (threadIdx.x == 0) {
        float t = 0.0f;
        #pragma unroll
        for (int w = 0; w < 8; ++w) t += wsum[w];
        cosp[blockIdx.x] = t;
    }
}

// ---------------------------------------------------------------------------
// Final deterministic combine.
// ---------------------------------------------------------------------------
__global__ __launch_bounds__(256) void final_kernel(const float* __restrict__ gram,
                                                    const float* __restrict__ cosp,
                                                    float* __restrict__ out) {
    __shared__ float red[256];
    __shared__ float totals[4];
    const int tid = threadIdx.x;

    for (int z = 0; z < 3; ++z) {
        float s = 0.0f;
        for (int i = tid; i < 4096; i += 256) s += gram[z * 4096 + i];
        red[tid] = s;
        __syncthreads();
        for (int off = 128; off > 0; off >>= 1) {
            if (tid < off) red[tid] += red[tid + off];
            __syncthreads();
        }
        if (tid == 0) totals[z] = red[0];
        __syncthreads();
    }
    {
        float s = 0.0f;
        for (int i = tid; i < 1024; i += 256) s += cosp[i];
        red[tid] = s;
        __syncthreads();
        for (int off = 128; off > 0; off >>= 1) {
            if (tid < off) red[tid] += red[tid + off];
            __syncthreads();
        }
        if (tid == 0) totals[3] = red[0];
        __syncthreads();
    }
    if (tid == 0) {
        const double denom = (double)B_ROWS * (double)(B_ROWS - 1);
        const double Sxx = (double)totals[0] / 3.0;
        const double Syy = (double)totals[1] / 3.0;
        const double Sxy = (double)totals[2] / 3.0;
        const double mmd = Sxx / denom + Syy / denom
                         - 2.0 * Sxy / ((double)B_ROWS * (double)B_ROWS);
        const double cosl = 1.0 - (double)totals[3] / (double)B_ROWS;
        out[0] = (float)(0.5 * mmd + 0.5 * cosl);
    }
}

// ---------------------------------------------------------------------------
// Launch
// ---------------------------------------------------------------------------
extern "C" void kernel_launch(void* const* d_in, const int* in_sizes, int n_in,
                              void* d_out, int out_size, void* d_ws, size_t ws_size,
                              hipStream_t stream) {
    const float* S = (const float*)d_in[0];
    const float* T = (const float*)d_in[1];

    // Workspace layout:
    //  [0, 8MB)   : Xbf  (8192x512 bf16)
    //  [8, 16MB)  : Ybf  (8192x512 bf16)
    //  [16MB ...) : f32 scratch
    unsigned short* Xbf = (unsigned short*)d_ws;
    unsigned short* Ybf = Xbf + (size_t)B_ROWS * D_DIM;
    float* w    = (float*)((char*)d_ws + (16u << 20));
    float* x2   = w;            // 8192
    float* y2   = w + 8192;     // 8192
    float* cent = w + 16384;    // 512
    float* cn   = w + 16896;    // 1
    float* pc   = w + 17408;    // 64*512 = 32768
    float* gram = w + 50176;    // 3*4096 = 12288
    float* cosp = w + 62464;    // 1024

    // 8192*512 / 8 floats-per-thread / 256 threads = 2048 blocks per matrix
    cvt_bf16_kernel<<<2048, 256, 0, stream>>>(S, (unsigned int*)Xbf);
    cvt_bf16_kernel<<<2048, 256, 0, stream>>>(T, (unsigned int*)Ybf);

    norms_kernel<<<2048, 256, 0, stream>>>(S, T, x2, y2);
    centroid_partial_kernel<<<64, 512, 0, stream>>>(T, pc);
    centroid_reduce_kernel<<<1, 512, 0, stream>>>(pc, cent, cn);
    gram_mmd_kernel<<<dim3(64, 64, 3), 256, 0, stream>>>(Xbf, Ybf, x2, y2, gram);
    cosine_kernel<<<1024, 256, 0, stream>>>(S, cent, x2, cn, cosp);
    final_kernel<<<1, 256, 0, stream>>>(gram, cosp, (float*)d_out);
}
